// EFFTrackHead_17875653886330
// MI455X (gfx1250) — matrix operand order repaired and checked by reading, not compiled, not run on hardware
//
#include <hip/hip_runtime.h>

#define FSZ  32
#define WIN  3
#define HID  160
#define HID4 640
#define NQ   16
#define NN_SAMPLES 4096

typedef __attribute__((ext_vector_type(16))) __bf16 bf16x16;
typedef __attribute__((ext_vector_type(8)))  float  v8f;

__device__ __forceinline__ unsigned short f2bf(float x) {
  union { float f; unsigned u; } v; v.f = x;
  unsigned r = v.u + 0x7FFFu + ((v.u >> 16) & 1u);   // round-to-nearest-even
  return (unsigned short)(r >> 16);
}

// ---------------- mean over Q (f32 -> bf16), for qe / fp ----------------
__global__ void mean_q_kernel(const float* __restrict__ in,
                              unsigned short* __restrict__ out, int total) {
  int idx = blockIdx.x * blockDim.x + threadIdx.x;
  if (idx >= total) return;
  int n = idx / HID, c = idx - n * HID;
  const float* p = in + (size_t)n * NQ * HID + c;
  float s = 0.f;
  #pragma unroll
  for (int q = 0; q < NQ; ++q) s += p[q * HID];
  out[idx] = f2bf(s * (1.0f / 16.0f));
}

// ---------------- pack weight [K,NN] f32 -> bf16 B-fragments ----------------
// layout: (((ng*KT + kt)*4 + t)*32 + lane)*16 + j
//   fragment: lane l -> col n = (ng*4+t)*16 + (l&15), element j -> k = kt*32 + (j&7) + 8*(l>>4) + 16*(j>>3)
// => for a wave with fixed ng, the entire B stream is base + kt*4096B + t*1024B (+16B)
__global__ void pack_w_kernel(const float* __restrict__ W,
                              unsigned short* __restrict__ out, int K, int NN) {
  int idx = blockIdx.x * blockDim.x + threadIdx.x;
  if (idx >= K * NN) return;
  int KT = K >> 5;
  int j    = idx & 15;
  int l    = (idx >> 4) & 31;
  int t    = (idx >> 9) & 3;
  int rest = idx >> 11;           // ng*KT + kt
  int kt   = rest % KT;
  int ng   = rest / KT;
  int k = kt * 32 + (j & 7) + 8 * (l >> 4) + 16 * (j >> 3);
  int n = (ng * 4 + t) * 16 + (l & 15);
  out[idx] = f2bf(W[(size_t)k * NN + n]);
}

// ---------------- bf16 WMMA GEMM: C[M,NN] = relu(A[M,K] @ Bpacked + bias) ----------------
// one wave -> 16x64 output tile; software-pipelined: fragments for kt+1 are
// loaded before the 4 WMMAs of kt issue, all with immediate byte offsets.
struct frag { union { bf16x16 v; uint4 q[2]; }; };

#define LOAD_A(dst, kt) do { \
  (dst).q[0] = *(const uint4*)(ap + (kt) * 32);      \
  (dst).q[1] = *(const uint4*)(ap + (kt) * 32 + 16); } while (0)

#define LOAD_B(dst, kt) do { \
  (dst)[0].q[0] = *(const uint4*)(bp + (kt) * 2048);        \
  (dst)[0].q[1] = *(const uint4*)(bp + (kt) * 2048 + 8);    \
  (dst)[1].q[0] = *(const uint4*)(bp + (kt) * 2048 + 512);  \
  (dst)[1].q[1] = *(const uint4*)(bp + (kt) * 2048 + 520);  \
  (dst)[2].q[0] = *(const uint4*)(bp + (kt) * 2048 + 1024); \
  (dst)[2].q[1] = *(const uint4*)(bp + (kt) * 2048 + 1032); \
  (dst)[3].q[0] = *(const uint4*)(bp + (kt) * 2048 + 1536); \
  (dst)[3].q[1] = *(const uint4*)(bp + (kt) * 2048 + 1544); } while (0)

#define MM4(af, bf) do { \
  acc[0] = __builtin_amdgcn_wmma_f32_16x16x32_bf16(false, (af).v, false, (bf)[0].v, (short)0, acc[0], false, false); \
  acc[1] = __builtin_amdgcn_wmma_f32_16x16x32_bf16(false, (af).v, false, (bf)[1].v, (short)0, acc[1], false, false); \
  acc[2] = __builtin_amdgcn_wmma_f32_16x16x32_bf16(false, (af).v, false, (bf)[2].v, (short)0, acc[2], false, false); \
  acc[3] = __builtin_amdgcn_wmma_f32_16x16x32_bf16(false, (af).v, false, (bf)[3].v, (short)0, acc[3], false, false); } while (0)

template <int KT>
__global__ __launch_bounds__(256) void gemm_bf16_kernel(
    const unsigned short* __restrict__ A, const unsigned short* __restrict__ Bp,
    const float* __restrict__ bias, unsigned short* __restrict__ C,
    int M, int NN, int relu) {
  const int K = KT * 32;
  int wid  = (blockIdx.x * blockDim.x + threadIdx.x) >> 5;
  int lane = threadIdx.x & 31;
  int nGroups = NN >> 6;
  int mt = wid / nGroups;
  int ng = wid - mt * nGroups;
  int m = lane & 15, h = lane >> 4;
  int row = mt * 16 + m;

  const __bf16* ap = (const __bf16*)A + (size_t)row * K + 8 * h;
  const __bf16* bp = (const __bf16*)Bp + (size_t)ng * KT * 2048 + lane * 16;

  v8f acc[4] = {};
  frag a, b[4], an, bn[4];
  LOAD_A(a, 0);
  LOAD_B(b, 0);
  #pragma unroll 2
  for (int kt = 0; kt < KT - 1; ++kt) {
    LOAD_A(an, kt + 1);        // prefetch next stage before consuming current
    LOAD_B(bn, kt + 1);
    MM4(a, b);
    a = an;
    b[0] = bn[0]; b[1] = bn[1]; b[2] = bn[2]; b[3] = bn[3];
  }
  MM4(a, b);

  // epilogue: C/D layout row = v + 8h, col = lane&15
  #pragma unroll
  for (int t = 0; t < 4; ++t) {
    int n = (ng * 4 + t) * 16 + (lane & 15);
    float bv = bias[n];
    #pragma unroll
    for (int v = 0; v < 8; ++v) {
      int r = mt * 16 + v + 8 * h;
      float x = acc[t][v] + bv;
      if (relu) x = fmaxf(x, 0.f);
      C[(size_t)r * NN + n] = f2bf(x);
    }
  }
}

// ---------------- tiny head: out[n, 0..O) = H[n,:] @ W4 + b4, one wave per sample ----------------
__global__ void head_dot_kernel(const unsigned short* __restrict__ H,
                                const float* __restrict__ W4, const float* __restrict__ b4,
                                float* __restrict__ out, int N, int K, int O, int ostride) {
  int gw = (blockIdx.x * blockDim.x + threadIdx.x) >> 5;
  int lane = threadIdx.x & 31;
  if (gw >= N) return;
  const __bf16* h = (const __bf16*)H + (size_t)gw * K;
  float p[4] = {0.f, 0.f, 0.f, 0.f};
  for (int k = lane; k < K; k += 32) {
    float hv = (float)h[k];
    for (int o = 0; o < O; ++o) p[o] += hv * W4[k * O + o];
  }
  for (int o = 0; o < O; ++o) {
    float v = p[o];
    for (int off = 16; off > 0; off >>= 1) v += __shfl_xor(v, off, 32);
    if (lane == 0) out[(size_t)gw * ostride + o] = v + b4[o];
  }
}

// ---------------- attn-map: Q-mean + blur/argmax + sigmoid window means ----------------
__global__ __launch_bounds__(128) void attn_kernel(const float* __restrict__ attn,
                                                   float* __restrict__ pp) {
  __shared__ float s_am[1024];
  __shared__ float s_sig[1024];
  __shared__ float s_rv[128];
  __shared__ int   s_ri[128];
  __shared__ float s_col[32];
  __shared__ float s_row[32];
  int n = blockIdx.x, tid = threadIdx.x;
  const float* base = attn + (size_t)n * NQ * 1024;
  #pragma unroll
  for (int i = 0; i < 8; ++i) {
    int e = tid + i * 128;
    float s = 0.f;
    #pragma unroll
    for (int q = 0; q < NQ; ++q) s += base[q * 1024 + e];
    float am = s * (1.f / 16.f);
    s_am[e] = am;
    float sg = 1.f / (1.f + expf(-am));
    s_sig[e] = fminf(fmaxf(sg, 1e-4f), 1.f - 1e-4f);
  }
  __syncthreads();
  // normalization is affine => argmax(blur) == argmax(normalized blur); skip mean/std
  const float e1 = 0.6065306597126334f;           // exp(-0.5)
  const float kc = 1.f / (1.f + 2.f * e1);
  const float kw[3] = {e1 * kc, kc, e1 * kc};
  float bv = -1e30f; int bi = 0;
  #pragma unroll
  for (int i = 0; i < 8; ++i) {
    int e = tid + i * 128;
    int r = e >> 5, c = e & 31;
    float acc = 0.f;
    #pragma unroll
    for (int dr = -1; dr <= 1; ++dr) {
      int rr = r + dr; rr = rr < 0 ? -rr : (rr > 31 ? 62 - rr : rr);   // reflect
      #pragma unroll
      for (int dc = -1; dc <= 1; ++dc) {
        int cc = c + dc; cc = cc < 0 ? -cc : (cc > 31 ? 62 - cc : cc);
        acc += kw[dr + 1] * kw[dc + 1] * s_am[rr * 32 + cc];
      }
    }
    if (acc > bv || (acc == bv && e < bi)) { bv = acc; bi = e; }
  }
  s_rv[tid] = bv; s_ri[tid] = bi;
  if (tid < 32) {
    float s = 0.f;
    for (int r = 0; r < 32; ++r) s += s_sig[r * 32 + tid];
    s_col[tid] = s;
  } else if (tid < 64) {
    int r = tid - 32; float s = 0.f;
    for (int c = 0; c < 32; ++c) s += s_sig[r * 32 + c];
    s_row[r] = s;
  }
  __syncthreads();
  if (tid == 0) {
    float best = s_rv[0]; int bidx = s_ri[0];
    for (int i = 1; i < 128; ++i)
      if (s_rv[i] > best || (s_rv[i] == best && s_ri[i] < bidx)) { best = s_rv[i]; bidx = s_ri[i]; }
    int px = bidx & 31, py = bidx >> 5;
    int clo = max(0, px - WIN), chi = min(FSZ, px + WIN);
    int rlo = max(0, py - WIN), rhi = min(FSZ, py + WIN);
    float cs = 0.f; for (int c = clo; c < chi; ++c) cs += s_col[c];
    float rs = 0.f; for (int r = rlo; r < rhi; ++r) rs += s_row[r];
    float tempx = cs / (32.f * (float)(chi - clo));
    float tempy = rs / (32.f * (float)(rhi - rlo));
    float* o = pp + (size_t)n * 4;
    o[0] = (float)px; o[1] = (float)py;
    o[2] = tempx * 32.f; o[3] = tempy * 32.f;
  }
}

// ---------------- final bbox combine ----------------
__global__ void combine_kernel(const float* __restrict__ bbox_info,
                               const float* __restrict__ pp,
                               float* __restrict__ out, int N) {
  int n = blockIdx.x * blockDim.x + threadIdx.x;
  if (n >= N) return;
  float so0 = fminf(fmaxf(bbox_info[n * 4 + 0], -2.f), 2.f);
  float so1 = fminf(fmaxf(bbox_info[n * 4 + 1], -2.f), 2.f);
  float po0 = fminf(fmaxf(bbox_info[n * 4 + 2], -2.f), 2.f);
  float po1 = fminf(fmaxf(bbox_info[n * 4 + 3], -2.f), 2.f);
  float px = pp[n * 4 + 0], py = pp[n * 4 + 1];
  float sx = pp[n * 4 + 2], sy = pp[n * 4 + 3];
  float w = (sx + so0) * (1.f / 32.f);
  float h = (sy + so1) * (1.f / 32.f);
  float x = (px + po0) * (1.f / 32.f) - 0.5f * w;
  float y = (py + po1) * (1.f / 32.f) - 0.5f * h;
  out[n * 4 + 0] = x; out[n * 4 + 1] = y; out[n * 4 + 2] = w; out[n * 4 + 3] = h;
}

extern "C" void kernel_launch(void* const* d_in, const int* in_sizes, int n_in,
                              void* d_out, int out_size, void* d_ws, size_t ws_size,
                              hipStream_t stream) {
  (void)in_sizes; (void)n_in; (void)out_size; (void)ws_size;
  const int N = NN_SAMPLES;
  const float* feature_pos = (const float*)d_in[0];
  const float* query_emb   = (const float*)d_in[1];
  const float* attn        = (const float*)d_in[2];
  const float* ws1 = (const float*)d_in[3];  const float* bs1 = (const float*)d_in[4];
  const float* ws2 = (const float*)d_in[5];  const float* bs2 = (const float*)d_in[6];
  const float* ws3 = (const float*)d_in[7];  const float* bs3 = (const float*)d_in[8];
  const float* ws4 = (const float*)d_in[9];  const float* bs4 = (const float*)d_in[10];
  const float* wp1 = (const float*)d_in[11]; const float* bp1 = (const float*)d_in[12];
  const float* wp2 = (const float*)d_in[13]; const float* bp2 = (const float*)d_in[14];
  const float* wp3 = (const float*)d_in[15]; const float* bp3 = (const float*)d_in[16];
  const float* wp4 = (const float*)d_in[17]; const float* bp4 = (const float*)d_in[18];
  float* out = (float*)d_out;

  char* wptr = (char*)d_ws;
  auto alloc = [&](size_t bytes) -> char* {
    char* p = wptr; wptr += (bytes + 255) & ~(size_t)255; return p;
  };
  unsigned short* qe_bf = (unsigned short*)alloc((size_t)N * HID * 2);
  unsigned short* fp_bf = (unsigned short*)alloc((size_t)N * HID * 2);
  unsigned short* Ha    = (unsigned short*)alloc((size_t)N * HID4 * 2);
  unsigned short* Hb    = (unsigned short*)alloc((size_t)N * HID4 * 2);
  unsigned short* Ws1p  = (unsigned short*)alloc((size_t)HID  * HID4 * 2);
  unsigned short* Ws2p  = (unsigned short*)alloc((size_t)HID4 * HID4 * 2);
  unsigned short* Ws3p  = (unsigned short*)alloc((size_t)HID4 * HID4 * 2);
  unsigned short* Wp1p  = (unsigned short*)alloc((size_t)HID  * HID4 * 2);
  unsigned short* Wp2p  = (unsigned short*)alloc((size_t)HID4 * HID4 * 2);
  unsigned short* Wp3p  = (unsigned short*)alloc((size_t)HID4 * HID4 * 2);
  float* attn_pp   = (float*)alloc((size_t)N * 4 * 4);
  float* bbox_info = (float*)alloc((size_t)N * 4 * 4);

  // Q-means -> bf16 activations
  mean_q_kernel<<<(N * HID + 255) / 256, 256, 0, stream>>>(query_emb, qe_bf, N * HID);
  mean_q_kernel<<<(N * HID + 255) / 256, 256, 0, stream>>>(feature_pos, fp_bf, N * HID);
  // weight packing (bf16 B-fragments, n-group-major streaming layout)
  pack_w_kernel<<<(HID  * HID4 + 255) / 256, 256, 0, stream>>>(ws1, Ws1p, HID,  HID4);
  pack_w_kernel<<<(HID4 * HID4 + 255) / 256, 256, 0, stream>>>(ws2, Ws2p, HID4, HID4);
  pack_w_kernel<<<(HID4 * HID4 + 255) / 256, 256, 0, stream>>>(ws3, Ws3p, HID4, HID4);
  pack_w_kernel<<<(HID  * HID4 + 255) / 256, 256, 0, stream>>>(wp1, Wp1p, HID,  HID4);
  pack_w_kernel<<<(HID4 * HID4 + 255) / 256, 256, 0, stream>>>(wp2, Wp2p, HID4, HID4);
  pack_w_kernel<<<(HID4 * HID4 + 255) / 256, 256, 0, stream>>>(wp3, Wp3p, HID4, HID4);
  // attn-map processing (fused Q-mean, single pass over 268 MB)
  attn_kernel<<<N, 128, 0, stream>>>(attn, attn_pp);

  // WMMA GEMMs: (M/16)*(NN/64) waves = 2560 -> 320 blocks of 256
  int gblocks = ((N / 16) * (HID4 / 64) * 32) / 256;
  // score MLP
  gemm_bf16_kernel<HID / 32><<<gblocks, 256, 0, stream>>>(qe_bf, Ws1p, bs1, Ha, N, HID4, 1);
  gemm_bf16_kernel<HID4 / 32><<<gblocks, 256, 0, stream>>>(Ha, Ws2p, bs2, Hb, N, HID4, 1);
  gemm_bf16_kernel<HID4 / 32><<<gblocks, 256, 0, stream>>>(Hb, Ws3p, bs3, Ha, N, HID4, 1);
  head_dot_kernel<<<(N * 32 + 255) / 256, 256, 0, stream>>>(Ha, ws4, bs4, out + (size_t)N * 4, N, HID4, 1, 1);
  // bbox MLP (ping-pong buffers, stream-ordered reuse)
  gemm_bf16_kernel<HID / 32><<<gblocks, 256, 0, stream>>>(fp_bf, Wp1p, bp1, Hb, N, HID4, 1);
  gemm_bf16_kernel<HID4 / 32><<<gblocks, 256, 0, stream>>>(Hb, Wp2p, bp2, Ha, N, HID4, 1);
  gemm_bf16_kernel<HID4 / 32><<<gblocks, 256, 0, stream>>>(Ha, Wp3p, bp3, Hb, N, HID4, 1);
  head_dot_kernel<<<(N * 32 + 255) / 256, 256, 0, stream>>>(Hb, wp4, bp4, bbox_info, N, HID4, 4, 4);

  combine_kernel<<<(N + 255) / 256, 256, 0, stream>>>(bbox_info, attn_pp, out, N);
}
